// LinearLayer_60387240182488
// MI455X (gfx1250) — compile-verified, matrix operand order checked by reading, not run on hardware
//
#include <hip/hip_runtime.h>
#include <hip/hip_bf16.h>
#include <stdint.h>

#define HID   96
#define INDIM 96
#define CDIM  192
#define BN_EPS 1e-5f
#define TPAD  200   // padded LDS row stride (bf16 elems): 100 dwords -> conflict-free

typedef __attribute__((ext_vector_type(16))) __bf16 v16bf;
typedef __attribute__((ext_vector_type(8)))  float  v8f;

// ---------------- helpers ----------------

__device__ __forceinline__ unsigned short f2bf(float f) {
    unsigned int u = __builtin_bit_cast(unsigned int, f);
    unsigned int r = u + 0x7FFFu + ((u >> 16) & 1u);   // round-to-nearest-even
    return (unsigned short)(r >> 16);
}

// 16-element bf16 fragment slice: lane needs K {base..base+7} and
// K {base+16..base+23} (caller already applied the +8 half-wave shift).
__device__ __forceinline__ v16bf load_frag(const unsigned short* p) {
    uint4 lo = *(const uint4*)(p);
    uint4 hi = *(const uint4*)(p + 16);
    union { unsigned int u[8]; v16bf v; } t;
    t.u[0] = lo.x; t.u[1] = lo.y; t.u[2] = lo.z; t.u[3] = lo.w;
    t.u[4] = hi.x; t.u[5] = hi.y; t.u[6] = hi.z; t.u[7] = hi.w;
    return t.v;
}

// ---------------- elementwise / scatter kernels ----------------

__global__ void k_concat_scatter(const float* __restrict__ hid,
                                 const float* __restrict__ qry,
                                 const int*   __restrict__ seg,
                                 unsigned short* __restrict__ hxb,
                                 float* __restrict__ vsum,
                                 float* __restrict__ cnt,
                                 int N) {
    int i = blockIdx.x * blockDim.x + threadIdx.x;
    int total = N * CDIM;
    if (i >= total) return;
    int p = i / CDIM, c = i - p * CDIM;
    float v = (c < HID) ? hid[p * HID + c] : qry[p * INDIM + (c - HID)];
    hxb[(size_t)p * CDIM + c] = f2bf(v);
    int s = seg[p];
    atomicAdd(&vsum[(size_t)s * CDIM + c], v);
    if (c == 0) atomicAdd(&cnt[s], 1.0f);
}

__global__ void k_voxel_mean(const float* __restrict__ vsum,
                             const float* __restrict__ cnt,
                             unsigned short* __restrict__ vfb,
                             int M) {
    int i = blockIdx.x * blockDim.x + threadIdx.x;
    int total = (M + 1) * CDIM;
    if (i >= total) return;
    int r = i / CDIM;
    float f = 0.0f;
    if (r < M) f = vsum[i] / fmaxf(cnt[r], 1.0f);
    vfb[i] = f2bf(f);
}

// W_conv [27,192,96] f32 -> Wt [27,96,192] bf16 (column-major per tap).
__global__ void k_wconv_prep(const float* __restrict__ W,
                             unsigned short* __restrict__ Wt) {
    int i = blockIdx.x * blockDim.x + threadIdx.x;
    int total = 27 * HID * CDIM;
    if (i >= total) return;
    int k = i / (HID * CDIM);
    int rem = i - k * HID * CDIM;
    int n = rem / CDIM;
    int c = rem - n * CDIM;
    Wt[i] = f2bf(W[((size_t)k * CDIM + c) * HID + n]);
}

// W_lin [192,96] f32 -> Wlt [96,192] bf16.
__global__ void k_wlin_prep(const float* __restrict__ W,
                            unsigned short* __restrict__ Wt) {
    int i = blockIdx.x * blockDim.x + threadIdx.x;
    int total = INDIM * CDIM;
    if (i >= total) return;
    int n = i / CDIM;
    int c = i - n * CDIM;
    Wt[i] = f2bf(W[(size_t)c * INDIM + n]);
}

// ---------------- WMMA sparse conv ----------------
// 256-thread block = 8 waves; each wave owns 3 M-tiles (block covers 24 tiles).
// Double-buffered LDS: async-stage tap k+1 (ASYNCcnt) while WMMAs chew tap k.
__global__ void __launch_bounds__(256)
k_sparse_conv_wmma(const unsigned short* __restrict__ vfb,
                   const unsigned short* __restrict__ Wt,
                   const int* __restrict__ nbr,
                   float* __restrict__ accbuf,
                   int M) {
    __shared__ unsigned short sB[2][HID * TPAD];   // 2 x 38400 B

    const int tid  = threadIdx.x;
    const int wv   = tid >> 5;
    const int lane = tid & 31;
    const int r    = lane & 15;
    const int half = lane >> 4;

    const int tbase = blockIdx.x * 24 + wv * 3;
    const int m0 = (tbase + 0) * 16 + r;
    const int m1 = (tbase + 1) * 16 + r;
    const int m2 = (tbase + 2) * 16 + r;

    v8f acc[3][6];
#pragma unroll
    for (int t = 0; t < 3; ++t)
#pragma unroll
        for (int n = 0; n < 6; ++n) acc[t][n] = v8f{};

    const unsigned long long wbase = (unsigned long long)(uintptr_t)Wt;

    auto stage = [&](int k, int buf) {
        // Wt[k]: 96 rows x 192 bf16 = 2304 b128 chunks, 9 per thread
#pragma unroll
        for (int c = 0; c < 9; ++c) {
            int chunk = tid + c * 256;
            int row   = chunk / 24;
            int col8  = chunk - row * 24;
            unsigned int ldsoff = (unsigned int)(uintptr_t)&sB[buf][row * TPAD + col8 * 8];
            unsigned int goff   = (unsigned int)(((k * HID + row) * CDIM + col8 * 8) * 2);
            asm volatile("global_load_async_to_lds_b128 %0, %1, %2"
                         :: "v"(ldsoff), "v"(goff), "s"(wbase) : "memory");
        }
    };

    stage(0, 0);
    for (int k = 0; k < 27; ++k) {
        asm volatile("s_wait_asynccnt 0x0" ::: "memory");   // tap k staged (this wave)
        __syncthreads();                                    // all waves staged + reads of buf^1 done
        if (k + 1 < 27) stage(k + 1, (k + 1) & 1);          // prefetch next tap (overlaps WMMAs)

        const unsigned short* sbuf = &sB[k & 1][0];
        int row0 = (m0 < M) ? nbr[m0 * 27 + k] : M;         // row M = zero pad
        int row1 = (m1 < M) ? nbr[m1 * 27 + k] : M;
        int row2 = (m2 < M) ? nbr[m2 * 27 + k] : M;
        const unsigned short* a0p = vfb + (size_t)row0 * CDIM + half * 8;
        const unsigned short* a1p = vfb + (size_t)row1 * CDIM + half * 8;
        const unsigned short* a2p = vfb + (size_t)row2 * CDIM + half * 8;
#pragma unroll
        for (int kc = 0; kc < 6; ++kc) {
            v16bf a0 = load_frag(a0p + kc * 32);
            v16bf a1 = load_frag(a1p + kc * 32);
            v16bf a2 = load_frag(a2p + kc * 32);
#pragma unroll
            for (int nt = 0; nt < 6; ++nt) {
                v16bf b = load_frag(sbuf + (nt * 16 + r) * TPAD + kc * 32 + half * 8);
                acc[0][nt] = __builtin_amdgcn_wmma_f32_16x16x32_bf16(
                    false, a0, false, b, (short)0, acc[0][nt], false, false);
                acc[1][nt] = __builtin_amdgcn_wmma_f32_16x16x32_bf16(
                    false, a1, false, b, (short)0, acc[1][nt], false, false);
                acc[2][nt] = __builtin_amdgcn_wmma_f32_16x16x32_bf16(
                    false, a2, false, b, (short)0, acc[2][nt], false, false);
            }
        }
    }
#pragma unroll
    for (int t = 0; t < 3; ++t) {
#pragma unroll
        for (int nt = 0; nt < 6; ++nt) {
#pragma unroll
            for (int i = 0; i < 8; ++i) {
                int g = (tbase + t) * 16 + i + half * 8;
                if (g < M) accbuf[(size_t)g * HID + nt * 16 + r] = acc[t][nt][i];
            }
        }
    }
}

// ---------------- BatchNorm ----------------

__global__ void k_bn_stats(const float* __restrict__ accbuf,
                           const float* __restrict__ gamma,
                           const float* __restrict__ beta,
                           float* __restrict__ scale,
                           float* __restrict__ shift,
                           int M) {
    __shared__ float ssum[256], ssq[256];
    int c = blockIdx.x;
    float s = 0.0f, q = 0.0f;
    for (int rr = threadIdx.x; rr < M; rr += blockDim.x) {
        float v = accbuf[(size_t)rr * HID + c];
        s += v; q += v * v;
    }
    ssum[threadIdx.x] = s; ssq[threadIdx.x] = q;
    __syncthreads();
    for (int off = 128; off > 0; off >>= 1) {
        if (threadIdx.x < off) {
            ssum[threadIdx.x] += ssum[threadIdx.x + off];
            ssq[threadIdx.x]  += ssq[threadIdx.x + off];
        }
        __syncthreads();
    }
    if (threadIdx.x == 0) {
        float mu  = ssum[0] / (float)M;
        float var = ssq[0] / (float)M - mu * mu;
        float sc  = gamma[c] * rsqrtf(var + BN_EPS);
        scale[c]  = sc;
        shift[c]  = beta[c] - mu * sc;
    }
}

__global__ void k_bn_relu(const float* __restrict__ accbuf,
                          const float* __restrict__ scale,
                          const float* __restrict__ shift,
                          float* __restrict__ y, int M) {
    int i = blockIdx.x * blockDim.x + threadIdx.x;
    int total = (M + 1) * HID;
    if (i >= total) return;
    int r = i / HID, c = i - r * HID;
    float v = 0.0f;
    if (r < M) v = fmaxf(accbuf[i] * scale[c] + shift[c], 0.0f);
    y[i] = v;
}

// ---------------- devoxelize + residual linear ----------------

__global__ void k_devox(const float* __restrict__ y,
                        const float* __restrict__ cw,
                        const int*   __restrict__ cidx,
                        float* __restrict__ out, int N) {
    int i = blockIdx.x * blockDim.x + threadIdx.x;
    int total = N * HID;
    if (i >= total) return;
    int p = i / HID, c = i - p * HID;
    float s = 0.0f;
#pragma unroll
    for (int j = 0; j < 8; ++j) {
        int idx = cidx[p * 8 + j];
        s += cw[p * 8 + j] * y[(size_t)idx * HID + c];
    }
    out[i] = s;
}

__global__ void k_linear_wmma(const unsigned short* __restrict__ hxb,
                              const unsigned short* __restrict__ Wlt,
                              const float* __restrict__ blin,
                              float* __restrict__ out,
                              int N, int Ntiles) {
    int wave = (blockIdx.x * blockDim.x + threadIdx.x) >> 5;
    int lane = threadIdx.x & 31;
    if (wave >= Ntiles) return;
    int p0   = wave * 16;
    int r    = lane & 15;
    int half = lane >> 4;

    v8f acc[6];
#pragma unroll
    for (int t = 0; t < 6; ++t) acc[t] = v8f{};

    const unsigned short* arow = hxb + (size_t)(p0 + r) * CDIM + half * 8;
#pragma unroll
    for (int kc = 0; kc < 6; ++kc) {
        v16bf a = load_frag(arow + kc * 32);
#pragma unroll
        for (int nt = 0; nt < 6; ++nt) {
            v16bf b = load_frag(Wlt + (size_t)(nt * 16 + r) * CDIM
                                    + kc * 32 + half * 8);
            acc[nt] = __builtin_amdgcn_wmma_f32_16x16x32_bf16(
                false, a, false, b, (short)0, acc[nt], false, false);
        }
    }
#pragma unroll
    for (int nt = 0; nt < 6; ++nt) {
#pragma unroll
        for (int i = 0; i < 8; ++i) {
            int gr = p0 + i + half * 8;
            if (gr < N) {
                int col = nt * 16 + r;
                size_t o = (size_t)gr * HID + col;
                out[o] = out[o] + acc[nt][i] + blin[col];
            }
        }
    }
}

// ---------------- launch ----------------

extern "C" void kernel_launch(void* const* d_in, const int* in_sizes, int n_in,
                              void* d_out, int out_size, void* d_ws, size_t ws_size,
                              hipStream_t stream) {
    const float* hid   = (const float*)d_in[0];
    const float* qry   = (const float*)d_in[1];
    const float* Wconv = (const float*)d_in[2];
    const float* gamma = (const float*)d_in[3];
    const float* beta  = (const float*)d_in[4];
    const float* Wlin  = (const float*)d_in[5];
    const float* blin  = (const float*)d_in[6];
    const float* cw    = (const float*)d_in[7];
    const int*   seg   = (const int*)d_in[8];
    const int*   nbr   = (const int*)d_in[9];
    const int*   cidx  = (const int*)d_in[10];
    float* out = (float*)d_out;

    const int N      = in_sizes[0] / HID;
    const int M      = in_sizes[9] / 27;
    const int Npad   = ((N + 15) / 16) * 16;
    const int Ntiles = Npad / 16;
    const int Mtiles = (M + 15) / 16;
    const int Mblks  = (Mtiles + 23) / 24;     // 24 tiles per block (8 waves x 3)

    char* ws = (char*)d_ws;
    size_t off = 0;
    auto alloc = [&](size_t bytes) {
        void* p = ws + off;
        off = (off + bytes + 255) & ~(size_t)255;
        return p;
    };
    unsigned short* hxb  = (unsigned short*)alloc((size_t)Npad * CDIM * 2);
    float*          vsum = (float*)alloc((size_t)(M + 1) * CDIM * 4);
    float*          cnt  = (float*)alloc((size_t)(M + 1) * 4);
    unsigned short* vfb  = (unsigned short*)alloc((size_t)(M + 1) * CDIM * 2);
    unsigned short* Wt   = (unsigned short*)alloc((size_t)27 * HID * CDIM * 2);
    unsigned short* Wlt  = (unsigned short*)alloc((size_t)INDIM * CDIM * 2);
    float*          accb = (float*)alloc((size_t)M * HID * 4);
    float*          scl  = (float*)alloc(96 * 4);
    float*          shf  = (float*)alloc(96 * 4);
    float*          ybuf = (float*)alloc((size_t)(M + 1) * HID * 4);

    const int B = 256;
    auto grid = [](long long n) { return (unsigned)((n + 255) / 256); };

    hipMemsetAsync(hxb,  0, (size_t)Npad * CDIM * 2, stream);
    hipMemsetAsync(vsum, 0, (size_t)(M + 1) * CDIM * 4, stream);
    hipMemsetAsync(cnt,  0, (size_t)(M + 1) * 4, stream);

    k_concat_scatter<<<grid((long long)N * CDIM), B, 0, stream>>>(
        hid, qry, seg, hxb, vsum, cnt, N);
    k_voxel_mean<<<grid((long long)(M + 1) * CDIM), B, 0, stream>>>(vsum, cnt, vfb, M);
    k_wconv_prep<<<grid(27LL * HID * CDIM), B, 0, stream>>>(Wconv, Wt);
    k_wlin_prep<<<grid((long long)INDIM * CDIM), B, 0, stream>>>(Wlin, Wlt);

    k_sparse_conv_wmma<<<Mblks, B, 0, stream>>>(vfb, Wt, nbr, accb, M);

    k_bn_stats<<<HID, B, 0, stream>>>(accb, gamma, beta, scl, shf, M);
    k_bn_relu<<<grid((long long)(M + 1) * HID), B, 0, stream>>>(accb, scl, shf, ybuf, M);

    k_devox<<<grid((long long)N * HID), B, 0, stream>>>(ybuf, cw, cidx, out, N);
    k_linear_wmma<<<grid((long long)Ntiles * 32), B, 0, stream>>>(
        hxb, Wlt, blin, out, N, Ntiles);
}
